// GraphSageNet_67645734912961
// MI455X (gfx1250) — compile-verified, matrix operand order checked by reading, not run on hardware
//
#include <hip/hip_runtime.h>

#define NN 131072
#define EE 2097152
#define HH 128
#define GG 8192
#define AA 28
#define LL 4

typedef __attribute__((ext_vector_type(16))) __bf16 v16bf;
typedef __attribute__((ext_vector_type(8)))  __bf16 v8bf;
typedef __attribute__((ext_vector_type(8)))  float  v8f;

__device__ __forceinline__ __bf16 f2bf(float f) {
  union { float f; unsigned u; } in; in.f = f;
  unsigned r = in.u + 0x7FFFu + ((in.u >> 16) & 1u);   // round-to-nearest-even
  union { unsigned short s; __bf16 b; } out; out.s = (unsigned short)(r >> 16);
  return out.b;
}

// ---------------- utility kernels ----------------

__global__ void zero_f32_kernel(float* __restrict__ p, long n) {
  long i = (long)blockIdx.x * blockDim.x + threadIdx.x;
  if (i < n) p[i] = 0.0f;
}

__global__ void count_deg_kernel(const int* __restrict__ dst, float* __restrict__ deg) {
  int e = blockIdx.x * blockDim.x + threadIdx.x;
  if (e < EE) atomicAdd(&deg[dst[e]], 1.0f);
}

__global__ void count_gcnt_kernel(const int* __restrict__ gid, float* __restrict__ gcnt) {
  int v = blockIdx.x * blockDim.x + threadIdx.x;
  if (v < NN) atomicAdd(&gcnt[gid[v]], 1.0f);
}

__global__ void invert_kernel(float* __restrict__ p, int n) {
  int i = blockIdx.x * blockDim.x + threadIdx.x;
  if (i < n) p[i] = 1.0f / fmaxf(p[i], 1.0f);
}

// w_combo = W_ro @ W_pred   [128]
__global__ void wcombo_kernel(const float* __restrict__ W_ro,
                              const float* __restrict__ W_pred,
                              float* __restrict__ wc) {
  int i = threadIdx.x;           // 128 threads
  float acc = 0.0f;
  #pragma unroll 8
  for (int o = 0; o < HH; ++o) acc += W_ro[i * HH + o] * W_pred[o];
  wc[i] = acc;
}

__global__ void out_init_kernel(float* __restrict__ out, const float* __restrict__ b_pred) {
  int g = blockIdx.x * blockDim.x + threadIdx.x;
  if (g < GG) out[g] = b_pred[0];
}

// pack W_node[l][K][col] (f32) into per-lane bf16 B fragments:
// wpack[(((l*8+t)*8+kb)*32+lane)*16 + i], K = kb*32 + (i/8)*16 + (lane/16)*8 + (i%8)
__global__ void pack_weights_kernel(const float* __restrict__ W_node,
                                    __bf16* __restrict__ wpack) {
  int idx = blockIdx.x * blockDim.x + threadIdx.x;   // 4*8*8*32*16 = 131072
  int i    = idx & 15;
  int lane = (idx >> 4) & 31;
  int kb   = (idx >> 9) & 7;
  int t    = (idx >> 12) & 7;
  int l    = idx >> 15;
  int K    = kb * 32 + (i >> 3) * 16 + (lane >> 4) * 8 + (i & 7);
  int col  = t * 16 + (lane & 15);
  wpack[idx] = f2bf(W_node[(l * 256 + K) * HH + col]);
}

// ---------------- pipeline kernels ----------------

// h0 = h_in @ W_emb : [N,28] x [28,128]
__global__ void __launch_bounds__(128) embed_kernel(const float* __restrict__ hin,
                                                    const float* __restrict__ Wemb,
                                                    float* __restrict__ h) {
  __shared__ float srow[AA];
  int node = blockIdx.x;
  int tid = threadIdx.x;
  if (tid < AA) srow[tid] = hin[node * AA + tid];
  __syncthreads();
  float acc = 0.0f;
  #pragma unroll
  for (int a = 0; a < AA; ++a) acc += srow[a] * Wemb[a * HH + tid];
  h[node * HH + tid] = acc;
}

// c[dst] += h[src] ; one wave per edge, lane covers 4 features stride-32
__global__ void __launch_bounds__(256) aggregate_kernel(const float* __restrict__ h,
                                                        const int* __restrict__ src,
                                                        const int* __restrict__ dst,
                                                        float* __restrict__ c) {
  int e = (blockIdx.x * blockDim.x + threadIdx.x) >> 5;
  int lane = threadIdx.x & 31;
  if (e < EE) {
    int s = src[e], d = dst[e];
    const float* hs = h + (long)s * HH;
    float* cd = c + (long)d * HH;
    #pragma unroll
    for (int f = lane; f < HH; f += 32) atomicAdd(&cd[f], hs[f]);
  }
}

// h = relu([h || c*inv_deg] @ W_node[l] + b) * snorm + h   (in-place, 16 rows/block)
__global__ void __launch_bounds__(256) node_apply_kernel(float* __restrict__ h,
                                                         const float* __restrict__ c,
                                                         const __bf16* __restrict__ wpack,
                                                         const float* __restrict__ b_node,
                                                         const float* __restrict__ snorm,
                                                         const float* __restrict__ inv_deg,
                                                         int layer) {
  __shared__ __bf16 smA[16 * 264];       // 16 rows x 256 (K) , padded stride 264
  const int node0 = blockIdx.x * 16;
  const int tid = threadIdx.x;

  // stage A = [h || c*inv_deg] as bf16
  #pragma unroll
  for (int j = 0; j < 16; ++j) {
    int idx  = tid + 256 * j;            // 0..4095
    int row  = idx >> 8;
    int col  = idx & 255;
    int node = node0 + row;
    float v = (col < HH) ? h[(long)node * HH + col]
                         : c[(long)node * HH + (col - HH)] * inv_deg[node];
    smA[row * 264 + col] = f2bf(v);
  }
  __syncthreads();

  const int wave = tid >> 5;             // output column tile (0..7)
  const int lane = tid & 31;
  const int half = lane >> 4;
  const int arow = lane & 15;
  v8f acc = {};

  const __bf16* bp = wpack + (((long)(layer * 8 + wave) * 8) * 32 + lane) * 16;
  const __bf16* ap = smA + arow * 264 + half * 8;

  #pragma unroll
  for (int kb = 0; kb < 8; ++kb) {
    v8bf lo = *(const v8bf*)(ap + kb * 32);
    v8bf hi = *(const v8bf*)(ap + kb * 32 + 16);
    v16bf a;
    #pragma unroll
    for (int i = 0; i < 8; ++i) { a[i] = lo[i]; a[8 + i] = hi[i]; }
    v16bf b = *(const v16bf*)(bp + kb * 512);   // 32 lanes * 16 elems per kb step
    acc = __builtin_amdgcn_wmma_f32_16x16x32_bf16(
        false, a, false, b, (short)0, acc, false, false);
  }

  int colg = wave * 16 + (lane & 15);
  float bias = b_node[layer * HH + colg];
  #pragma unroll
  for (int r = 0; r < 8; ++r) {
    int row  = r + 8 * half;
    int node = node0 + row;
    float v = acc[r] + bias;
    v = fmaxf(v, 0.0f);
    v = v * snorm[node] + h[(long)node * HH + colg];   // graph_norm + residual
    h[(long)node * HH + colg] = v;
  }
}

// out[g] += inv_gcnt[g] * dot(h[node], w_combo) ; one wave per node
__global__ void __launch_bounds__(256) readout_kernel(const float* __restrict__ h,
                                                      const float* __restrict__ wc,
                                                      const int* __restrict__ gid,
                                                      const float* __restrict__ inv_gcnt,
                                                      float* __restrict__ out) {
  int node = (blockIdx.x * blockDim.x + threadIdx.x) >> 5;
  int lane = threadIdx.x & 31;
  if (node < NN) {
    const float* hr = h + (long)node * HH;
    float s = 0.0f;
    #pragma unroll
    for (int f = lane; f < HH; f += 32) s += hr[f] * wc[f];
    s += __shfl_xor(s, 1);
    s += __shfl_xor(s, 2);
    s += __shfl_xor(s, 4);
    s += __shfl_xor(s, 8);
    s += __shfl_xor(s, 16);
    if (lane == 0) {
      int g = gid[node];
      atomicAdd(&out[g], s * inv_gcnt[g]);
    }
  }
}

// ---------------- launch ----------------

static inline size_t alignup(size_t x) { return (x + 255) & ~(size_t)255; }

extern "C" void kernel_launch(void* const* d_in, const int* in_sizes, int n_in,
                              void* d_out, int out_size, void* d_ws, size_t ws_size,
                              hipStream_t stream) {
  (void)in_sizes; (void)n_in; (void)out_size; (void)ws_size;
  const float* hin     = (const float*)d_in[0];   // [N,28]
  const float* snorm   = (const float*)d_in[1];   // [N,1]
  const float* W_emb   = (const float*)d_in[2];   // [28,128]
  const float* W_node  = (const float*)d_in[3];   // [4,256,128]
  const float* b_node  = (const float*)d_in[4];   // [4,128]
  const float* W_ro    = (const float*)d_in[5];   // [128,128]
  const float* W_pred  = (const float*)d_in[6];   // [128,1]
  const float* b_pred  = (const float*)d_in[7];   // [1]
  const int*   src     = (const int*)d_in[8];
  const int*   dst     = (const int*)d_in[9];
  const int*   gid     = (const int*)d_in[10];
  float* out = (float*)d_out;                     // [G,1]

  char* w = (char*)d_ws;
  float*  h        = (float*)w;  w += alignup((size_t)NN * HH * 4);
  float*  c        = (float*)w;  w += alignup((size_t)NN * HH * 4);
  float*  inv_deg  = (float*)w;  w += alignup((size_t)NN * 4);
  float*  inv_gcnt = (float*)w;  w += alignup((size_t)GG * 4);
  float*  wcombo   = (float*)w;  w += alignup((size_t)HH * 4);
  __bf16* wpack    = (__bf16*)w; w += alignup((size_t)LL * 8 * 8 * 32 * 16 * 2);

  // degrees & graph counts
  zero_f32_kernel<<<(NN + 255) / 256, 256, 0, stream>>>(inv_deg, NN);
  zero_f32_kernel<<<(GG + 255) / 256, 256, 0, stream>>>(inv_gcnt, GG);
  count_deg_kernel<<<(EE + 255) / 256, 256, 0, stream>>>(dst, inv_deg);
  count_gcnt_kernel<<<(NN + 255) / 256, 256, 0, stream>>>(gid, inv_gcnt);
  invert_kernel<<<(NN + 255) / 256, 256, 0, stream>>>(inv_deg, NN);
  invert_kernel<<<(GG + 255) / 256, 256, 0, stream>>>(inv_gcnt, GG);

  // weight prep
  pack_weights_kernel<<<(LL * 8 * 8 * 32 * 16) / 256, 256, 0, stream>>>(W_node, wpack);
  wcombo_kernel<<<1, 128, 0, stream>>>(W_ro, W_pred, wcombo);

  // embedding
  embed_kernel<<<NN, 128, 0, stream>>>(hin, W_emb, h);

  // layers
  for (int l = 0; l < LL; ++l) {
    zero_f32_kernel<<<((long)NN * HH + 255) / 256, 256, 0, stream>>>(c, (long)NN * HH);
    aggregate_kernel<<<(EE * 32) / 256, 256, 0, stream>>>(h, src, dst, c);
    node_apply_kernel<<<NN / 16, 256, 0, stream>>>(h, c, wpack, b_node, snorm, inv_deg, l);
  }

  // fused readout + predict
  out_init_kernel<<<(GG + 255) / 256, 256, 0, stream>>>(out, b_pred);
  readout_kernel<<<(NN * 32) / 256, 256, 0, stream>>>(h, wcombo, gid, inv_gcnt, out);
}
